// Model_84421877170228
// MI455X (gfx1250) — matrix operand order repaired and checked by reading, not compiled, not run on hardware
//
#include <hip/hip_runtime.h>

typedef _Float16 h8  __attribute__((ext_vector_type(8)));
typedef _Float16 h16 __attribute__((ext_vector_type(16)));
typedef float    f8  __attribute__((ext_vector_type(8)));

#define BB 32
#define NN 4096

static __device__ __forceinline__ float relu(float v) { return v > 0.0f ? v : 0.0f; }

// ---------------- [B,3,N] -> [B,N,3] ----------------
__global__ void k_transpose(const float* __restrict__ xyz, float* __restrict__ x0) {
    long i = (long)blockIdx.x * blockDim.x + threadIdx.x;
    if (i >= (long)BB * NN * 3) return;
    int d = (int)(i % 3);
    long t = i / 3;
    int n = (int)(t % NN);
    int b = (int)(t / NN);
    x0[i] = xyz[((long)b * 3 + d) * NN + n];
}

// ---------------- farthest point sampling: 1 block per batch ----------------
__global__ void k_fps(const float* __restrict__ pts, int Np, int S, int* __restrict__ out) {
    __shared__ float dist[4096];
    __shared__ float rv[256];
    __shared__ int   ri[256];
    __shared__ int   s_far;
    int b = blockIdx.x, t = threadIdx.x;
    const float* P = pts + (long)b * Np * 3;
    for (int n = t; n < Np; n += 256) dist[n] = 1e10f;
    if (t == 0) s_far = 0;
    __syncthreads();
    for (int s = 0; s < S; ++s) {
        int far = s_far;
        if (t == 0) out[b * S + s] = far;
        float cx = P[far * 3 + 0], cy = P[far * 3 + 1], cz = P[far * 3 + 2];
        float bv = -1.0f; int bi = 0;
        for (int n = t; n < Np; n += 256) {
            float dx = P[n * 3 + 0] - cx, dy = P[n * 3 + 1] - cy, dz = P[n * 3 + 2] - cz;
            float d = dx * dx + dy * dy + dz * dz;
            float nd = fminf(dist[n], d);
            dist[n] = nd;
            if (nd > bv) { bv = nd; bi = n; }     // ascending n => keeps smallest idx on ties
        }
        rv[t] = bv; ri[t] = bi;
        __syncthreads();
        for (int o = 128; o > 0; o >>= 1) {
            if (t < o) {
                if (rv[t + o] > rv[t] || (rv[t + o] == rv[t] && ri[t + o] < ri[t])) {
                    rv[t] = rv[t + o]; ri[t] = ri[t + o];
                }
            }
            __syncthreads();
        }
        if (t == 0) s_far = ri[0];
        __syncthreads();
    }
}

// ---------------- gather xyz rows by index ----------------
__global__ void k_gather3(const float* __restrict__ pts, const int* __restrict__ idx,
                          float* __restrict__ out, int Np, int S) {
    long i = (long)blockIdx.x * blockDim.x + threadIdx.x;
    if (i >= (long)BB * S * 3) return;
    int d = (int)(i % 3);
    long t = i / 3;
    int s = (int)(t % S);
    int b = (int)(t / S);
    out[i] = pts[((long)b * Np + idx[b * S + s]) * 3 + d];
}

// ---------------- ball query: 1 wave (32 lanes) per query point ----------------
__global__ void k_ball(const float* __restrict__ xyz, const float* __restrict__ cent,
                       int Np, int S, int K, float r2, int* __restrict__ gidx) {
    long gtid = (long)blockIdx.x * blockDim.x + threadIdx.x;
    int w = (int)(gtid >> 5);
    int lane = (int)(gtid & 31);
    if (w >= BB * S) return;
    int b = w / S;
    const float* P = xyz + (long)b * Np * 3;
    float cx = cent[w * 3 + 0], cy = cent[w * 3 + 1], cz = cent[w * 3 + 2];
    int* out = gidx + (long)w * K;
    int cnt = 0, first = 0;
    bool haveFirst = false;
    for (int base = 0; base < Np && cnt < K; base += 32) {
        int n = base + lane;
        bool pred = false;
        if (n < Np) {
            float dx = P[n * 3 + 0] - cx, dy = P[n * 3 + 1] - cy, dz = P[n * 3 + 2] - cz;
            pred = (dx * dx + dy * dy + dz * dz) <= r2;
        }
        unsigned long long mask = __ballot(pred);
        if (mask) {
            if (!haveFirst) { first = base + __ffsll(mask) - 1; haveFirst = true; }
            if (pred) {
                int pos = cnt + __popcll(mask & ((1ull << lane) - 1ull));
                if (pos < K) out[pos] = n;
            }
            cnt += (int)__popcll(mask);
        }
    }
    if (cnt > K) cnt = K;
    for (int p = cnt + lane; p < K; p += 32) out[p] = first;
}

// ---------------- group (xyz - center, concat feat), cast f16, K-pad ----------------
__global__ void k_group(const float* __restrict__ xyz, const float* __restrict__ cent,
                        const int* __restrict__ gidx, const float* __restrict__ feat,
                        int Np, int S, int K, int Cf, int Kpad, _Float16* __restrict__ X,
                        long total) {
    long i = (long)blockIdx.x * blockDim.x + threadIdx.x;
    if (i >= total) return;
    int c = (int)(i % Kpad);
    long r = i / Kpad;
    int k = (int)(r % K);
    long bs = r / K;
    int b = (int)(bs / S);
    int idx = gidx[bs * K + k];
    float v = 0.0f;
    if (c < 3) {
        v = xyz[((long)b * Np + idx) * 3 + c] - cent[bs * 3 + c];
    } else if (c < 3 + Cf) {
        v = feat[((long)b * Np + idx) * Cf + (c - 3)];
    }
    X[i] = (_Float16)v;
}

// ---------------- SA3: concat [xyz, feat], cast f16, K-pad ----------------
__global__ void k_concat_cast(const float* __restrict__ xyz, const float* __restrict__ feat,
                              int Cf, int Kpad, _Float16* __restrict__ X, long total) {
    long i = (long)blockIdx.x * blockDim.x + threadIdx.x;
    if (i >= total) return;
    int c = (int)(i % Kpad);
    long r = i / Kpad;
    float v = 0.0f;
    if (c < 3) v = xyz[r * 3 + c];
    else if (c < 3 + Cf) v = feat[r * Cf + (c - 3)];
    X[i] = (_Float16)v;
}

// ---------------- weight prep: W[Cin,Cout] f32 -> Wt[Npad,Kpad] f16 (transposed) ----------------
__global__ void k_prep_w(const float* __restrict__ W, int Cin, int Cout, int Kpad, int Npad,
                         _Float16* __restrict__ Wt) {
    long i = (long)blockIdx.x * blockDim.x + threadIdx.x;
    if (i >= (long)Npad * Kpad) return;
    int k = (int)(i % Kpad);
    int n = (int)(i / Kpad);
    float v = (n < Cout && k < Cin) ? W[(long)k * Cout + n] : 0.0f;
    Wt[i] = (_Float16)v;
}

// ---------------- WMMA GEMM: Y[M,Cout] = X[M,Kpad](f16) @ Wt[Npad,Kpad]^T (+bias) ----------------
// one wave per 32x32 output patch: 2x2 register blocking over 16x16 WMMA tiles.
// A fragment reused across 2 N-tiles, B fragment across 2 M-tiles -> 8 B128 loads / 4 wmmas.
// Requires M % 32 == 0 and Npad % 32 == 0 (guaranteed by the host driver).
__global__ void k_gemm(const _Float16* __restrict__ X, const _Float16* __restrict__ Wt,
                       const float* __restrict__ bias, float* __restrict__ Y,
                       int M, int Cout, int Npad, int Kpad) {
    int lane = threadIdx.x & 31;
    long waveId = ((long)blockIdx.x * blockDim.x + threadIdx.x) >> 5;
    int nbN = Npad >> 5;
    long total = (long)(M >> 5) * nbN;
    if (waveId >= total) return;             // wave-uniform: EXEC stays all-ones
    int mb = (int)(waveId / nbN);
    int nb = (int)(waveId % nbN);
    int hi = lane >> 4;                       // lane group 0/1
    int l15 = lane & 15;
    // A: lanes 0-15 row m, halves0-7 = K 0..7, halves8-15 = K 16..23; lanes 16-31 K+8
    const _Float16* pa0 = X + (long)(mb * 32 + l15) * Kpad + (hi ? 8 : 0);
    const _Float16* pa1 = pa0 + (long)16 * Kpad;
    // B: lane holds column n = lane&15; halves0-7 = K 0..7, halves8-15 = K 16..23; lanes 16-31 K+8
    const _Float16* pb0 = Wt + (long)(nb * 32 + l15) * Kpad + (hi ? 8 : 0);
    const _Float16* pb1 = pb0 + (long)16 * Kpad;
    f8 acc00 = {}, acc01 = {}, acc10 = {}, acc11 = {};
    for (int k0 = 0; k0 < Kpad; k0 += 32) {
        h8 a0lo = *(const h8*)(pa0 + k0);
        h8 a0hi = *(const h8*)(pa0 + k0 + 16);
        h8 a1lo = *(const h8*)(pa1 + k0);
        h8 a1hi = *(const h8*)(pa1 + k0 + 16);
        h16 a0 = __builtin_shufflevector(a0lo, a0hi, 0, 1, 2, 3, 4, 5, 6, 7,
                                         8, 9, 10, 11, 12, 13, 14, 15);
        h16 a1 = __builtin_shufflevector(a1lo, a1hi, 0, 1, 2, 3, 4, 5, 6, 7,
                                         8, 9, 10, 11, 12, 13, 14, 15);
        h16 b0 = __builtin_shufflevector(*(const h8*)(pb0 + k0), *(const h8*)(pb0 + k0 + 16),
                                         0, 1, 2, 3, 4, 5, 6, 7, 8, 9, 10, 11, 12, 13, 14, 15);
        h16 b1 = __builtin_shufflevector(*(const h8*)(pb1 + k0), *(const h8*)(pb1 + k0 + 16),
                                         0, 1, 2, 3, 4, 5, 6, 7, 8, 9, 10, 11, 12, 13, 14, 15);
        acc00 = __builtin_amdgcn_wmma_f32_16x16x32_f16(false, a0, false, b0,
                                                       (short)0, acc00, false, false);
        acc01 = __builtin_amdgcn_wmma_f32_16x16x32_f16(false, a0, false, b1,
                                                       (short)0, acc01, false, false);
        acc10 = __builtin_amdgcn_wmma_f32_16x16x32_f16(false, a1, false, b0,
                                                       (short)0, acc10, false, false);
        acc11 = __builtin_amdgcn_wmma_f32_16x16x32_f16(false, a1, false, b1,
                                                       (short)0, acc11, false, false);
    }
    // D layout: VGPR j -> row (tileRow + j + hi*8), col (tileCol + l15)
    int col0 = nb * 32 + l15;
    int col1 = col0 + 16;
    int r0 = mb * 32 + (hi ? 8 : 0);
    int r1 = r0 + 16;
    if (col0 < Cout) {
        float bv = bias ? bias[col0] : 0.0f;
#pragma unroll
        for (int j = 0; j < 8; ++j) {
            Y[(long)(r0 + j) * Cout + col0] = acc00[j] + bv;
            Y[(long)(r1 + j) * Cout + col0] = acc10[j] + bv;
        }
    }
    if (col1 < Cout) {
        float bv = bias ? bias[col1] : 0.0f;
#pragma unroll
        for (int j = 0; j < 8; ++j) {
            Y[(long)(r0 + j) * Cout + col1] = acc01[j] + bv;
            Y[(long)(r1 + j) * Cout + col1] = acc11[j] + bv;
        }
    }
}

// ---------------- per-channel batch stats: mean + rsqrt(var+eps) ----------------
__global__ void k_stats(const float* __restrict__ Y, int M, int C,
                        float* __restrict__ mean, float* __restrict__ rstd) {
    __shared__ float sh[256], sh2[256];
    int c = blockIdx.x, t = threadIdx.x;
    float s = 0.0f, s2 = 0.0f;
    for (int i = t; i < M; i += 256) {
        float v = Y[(long)i * C + c];
        s += v; s2 += v * v;
    }
    sh[t] = s; sh2[t] = s2;
    __syncthreads();
    for (int o = 128; o > 0; o >>= 1) {
        if (t < o) { sh[t] += sh[t + o]; sh2[t] += sh2[t + o]; }
        __syncthreads();
    }
    if (t == 0) {
        float m = sh[0] / (float)M;
        float var = sh2[0] / (float)M - m * m;
        mean[c] = m;
        rstd[c] = rsqrtf(var + 1e-5f);
    }
}

// ---------------- BN + ReLU + f16 cast (K-padded) ----------------
__global__ void k_norm_relu_cast(const float* __restrict__ Y, const float* __restrict__ mean,
                                 const float* __restrict__ rstd, const float* __restrict__ g,
                                 const float* __restrict__ beta, int C, int Kpad,
                                 _Float16* __restrict__ Xo, long total) {
    long i = (long)blockIdx.x * blockDim.x + threadIdx.x;
    if (i >= total) return;
    int c = (int)(i % Kpad);
    long r = i / Kpad;
    float v = 0.0f;
    if (c < C) {
        float y = Y[r * C + c];
        v = relu((y - mean[c]) * rstd[c] * g[c] + beta[c]);
    }
    Xo[i] = (_Float16)v;
}

// ---------------- BN + ReLU + max over K points of a group ----------------
__global__ void k_norm_relu_maxpool(const float* __restrict__ Y, const float* __restrict__ mean,
                                    const float* __restrict__ rstd, const float* __restrict__ g,
                                    const float* __restrict__ beta, int K, int C,
                                    float* __restrict__ out, long total) {
    long i = (long)blockIdx.x * blockDim.x + threadIdx.x;
    if (i >= total) return;
    int c = (int)(i % C);
    long grp = i / C;
    const float* p = Y + grp * (long)K * C + c;
    float m = mean[c], rs = rstd[c], gg = g[c], bb = beta[c];
    float best = 0.0f;                         // relu output >= 0
    for (int k = 0; k < K; ++k) {
        float v = relu((p[(long)k * C] - m) * rs * gg + bb);
        best = fmaxf(best, v);
    }
    out[grp * C + c] = best;
}

// ---------------- plain f32 -> f16 cast ----------------
__global__ void k_cast_f16(const float* __restrict__ in, _Float16* __restrict__ out, long n) {
    long i = (long)blockIdx.x * blockDim.x + threadIdx.x;
    if (i < n) out[i] = (_Float16)in[i];
}

// ============================================================================
// host side
// ============================================================================
struct ParamMap {
    int xyz;
    int saW[3][3], saG[3][3], saB[3][3];
    int fc1w, bn1g, bn1b, fc2w, bn2g, bn2b, fc3w, fc3b;
};

static const int g_cin[3][3]  = { {3, 64, 64},   {131, 128, 128}, {259, 256, 512}  };
static const int g_cout[3][3] = { {64, 64, 128}, {128, 128, 256}, {256, 512, 1024} };
static const int g_kpad[3][3] = { {32, 64, 64},  {160, 128, 128}, {288, 256, 512}  };

static void build_map(int conv, int* sz, ParamMap& m) {
    int p = 0;
#define PUT(v) (sz[p] = (v), p++)
    if (conv == 0) {  // insertion order: xyz, sa1..sa3 ({w,b,g,beta}), fc1, bn1, fc2, bn2, fc3
        m.xyz = PUT(BB * 3 * NN);
        for (int s = 0; s < 3; s++)
            for (int l = 0; l < 3; l++) {
                m.saW[s][l] = PUT(g_cin[s][l] * g_cout[s][l]);
                PUT(g_cout[s][l]);               // bias (cancels under BN)
                m.saG[s][l] = PUT(g_cout[s][l]);
                m.saB[s][l] = PUT(g_cout[s][l]);
            }
        m.fc1w = PUT(1024 * 512); PUT(512);
        m.bn1g = PUT(512); m.bn1b = PUT(512);
        m.fc2w = PUT(512 * 256); PUT(256);
        m.bn2g = PUT(256); m.bn2b = PUT(256);
        m.fc3w = PUT(256 * 40); m.fc3b = PUT(40);
    } else {          // sorted dict keys (jax tree_leaves); conv1: params then xyz; conv2: xyz first
        if (conv == 2) m.xyz = PUT(BB * 3 * NN);
        m.bn1b = PUT(512); m.bn1g = PUT(512);
        m.bn2b = PUT(256); m.bn2g = PUT(256);
        PUT(512); m.fc1w = PUT(1024 * 512);
        PUT(256); m.fc2w = PUT(512 * 256);
        m.fc3b = PUT(40); m.fc3w = PUT(256 * 40);
        for (int s = 0; s < 3; s++)
            for (int l = 0; l < 3; l++) {
                PUT(g_cout[s][l]);                   // b
                m.saB[s][l] = PUT(g_cout[s][l]);     // beta
                m.saG[s][l] = PUT(g_cout[s][l]);     // g
                m.saW[s][l] = PUT(g_cin[s][l] * g_cout[s][l]);
            }
        if (conv == 1) m.xyz = PUT(BB * 3 * NN);
    }
#undef PUT
}

static inline unsigned GD(long total, int block) {
    return (unsigned)((total + block - 1) / block);
}

extern "C" void kernel_launch(void* const* d_in, const int* in_sizes, int n_in,
                              void* d_out, int out_size, void* d_ws, size_t ws_size,
                              hipStream_t stream) {
    if (n_in < 47) return;
    // ---- detect input flattening convention ----
    ParamMap pm{};
    int chosen = 0;
    for (int conv : {0, 2, 1}) {
        int sz[47]; ParamMap m{};
        build_map(conv, sz, m);
        bool ok = true;
        for (int i = 0; i < 47; i++) if (sz[i] != in_sizes[i]) { ok = false; break; }
        if (ok) { pm = m; chosen = conv; break; }
        if (conv == 0) pm = m;  // fallback
    }
    (void)chosen;

    const float* IN[47];
    for (int i = 0; i < 47; i++) IN[i] = (const float*)d_in[i];

    // ---- workspace layout ----
    size_t off = 0;
    auto walloc = [&](size_t bytes) { size_t o = off; off = (off + bytes + 255) & ~(size_t)255; return o; };
    char* W8 = (char*)d_ws;
    size_t o_x0   = walloc((size_t)BB * NN * 3 * 4);
    size_t o_nx1  = walloc((size_t)BB * 512 * 3 * 4);
    size_t o_l1f  = walloc((size_t)BB * 512 * 128 * 4);
    size_t o_l2x  = walloc((size_t)BB * 128 * 3 * 4);
    size_t o_l2f  = walloc((size_t)BB * 128 * 256 * 4);
    size_t o_feat = walloc((size_t)BB * 1024 * 4);
    size_t o_mean = walloc(1024 * 4);
    size_t o_rstd = walloc(1024 * 4);
    size_t o_fps1 = walloc((size_t)BB * 512 * 4);
    size_t o_fps2 = walloc((size_t)BB * 128 * 4);
    size_t o_g1   = walloc((size_t)BB * 512 * 32 * 4);
    size_t o_g2   = walloc((size_t)BB * 128 * 64 * 4);
    size_t o_wt   = walloc((size_t)524288 * 2);
    size_t o_X    = walloc((size_t)262144 * 160 * 2);    // max f16 activation
    size_t o_Y    = walloc((size_t)67108864 * 4);        // max f32 pre-BN activation
    if (off > ws_size) return;  // workspace too small; bail deterministically

    float*    x0   = (float*)(W8 + o_x0);
    float*    nx1  = (float*)(W8 + o_nx1);
    float*    l1f  = (float*)(W8 + o_l1f);
    float*    l2x  = (float*)(W8 + o_l2x);
    float*    l2f  = (float*)(W8 + o_l2f);
    float*    feat = (float*)(W8 + o_feat);
    float*    mean = (float*)(W8 + o_mean);
    float*    rstd = (float*)(W8 + o_rstd);
    int*      fps1 = (int*)(W8 + o_fps1);
    int*      fps2 = (int*)(W8 + o_fps2);
    int*      g1   = (int*)(W8 + o_g1);
    int*      g2   = (int*)(W8 + o_g2);
    _Float16* wt   = (_Float16*)(W8 + o_wt);
    _Float16* X    = (_Float16*)(W8 + o_X);
    float*    Y    = (float*)(W8 + o_Y);

    // Npad must be a multiple of 32 for the 2x2-blocked WMMA GEMM.
    auto gemm = [&](const _Float16* Xi, const float* Wf, const float* bias, float* Yo,
                    int M, int Cin, int Cout, int Npad, int Kpad) {
        k_prep_w<<<GD((long)Npad * Kpad, 256), 256, 0, stream>>>(Wf, Cin, Cout, Kpad, Npad, wt);
        long waves = (long)(M / 32) * (Npad / 32);
        k_gemm<<<GD(waves * 32, 256), 256, 0, stream>>>(Xi, wt, bias, Yo, M, Cout, Npad, Kpad);
    };

    // ---- prologue: transpose ----
    k_transpose<<<GD((long)BB * NN * 3, 256), 256, 0, stream>>>(IN[pm.xyz], x0);

    // ---- SA1 ----
    {
        const int S = 512, K = 32, M = BB * S * K;
        k_fps<<<BB, 256, 0, stream>>>(x0, NN, S, fps1);
        k_gather3<<<GD((long)BB * S * 3, 256), 256, 0, stream>>>(x0, fps1, nx1, NN, S);
        k_ball<<<GD((long)BB * S * 32, 256), 256, 0, stream>>>(x0, nx1, NN, S, K, 0.04f, g1);
        long tot = (long)M * g_kpad[0][0];
        k_group<<<GD(tot, 256), 256, 0, stream>>>(x0, nx1, g1, nullptr, NN, S, K, 0,
                                                  g_kpad[0][0], X, tot);
        for (int l = 0; l < 3; l++) {
            int Cin = g_cin[0][l], Co = g_cout[0][l], Kp = g_kpad[0][l];
            gemm(X, IN[pm.saW[0][l]], nullptr, Y, M, Cin, Co, Co, Kp);
            k_stats<<<Co, 256, 0, stream>>>(Y, M, Co, mean, rstd);
            if (l < 2) {
                int Kpn = g_kpad[0][l + 1];
                long t2 = (long)M * Kpn;
                k_norm_relu_cast<<<GD(t2, 256), 256, 0, stream>>>(Y, mean, rstd,
                    IN[pm.saG[0][l]], IN[pm.saB[0][l]], Co, Kpn, X, t2);
            } else {
                long t2 = (long)BB * S * Co;
                k_norm_relu_maxpool<<<GD(t2, 256), 256, 0, stream>>>(Y, mean, rstd,
                    IN[pm.saG[0][l]], IN[pm.saB[0][l]], K, Co, l1f, t2);
            }
        }
    }

    // ---- SA2 ----
    {
        const int S = 128, K = 64, M = BB * S * K;
        k_fps<<<BB, 256, 0, stream>>>(nx1, 512, S, fps2);
        k_gather3<<<GD((long)BB * S * 3, 256), 256, 0, stream>>>(nx1, fps2, l2x, 512, S);
        k_ball<<<GD((long)BB * S * 32, 256), 256, 0, stream>>>(nx1, l2x, 512, S, K, 0.16f, g2);
        long tot = (long)M * g_kpad[1][0];
        k_group<<<GD(tot, 256), 256, 0, stream>>>(nx1, l2x, g2, l1f, 512, S, K, 128,
                                                  g_kpad[1][0], X, tot);
        for (int l = 0; l < 3; l++) {
            int Cin = g_cin[1][l], Co = g_cout[1][l], Kp = g_kpad[1][l];
            gemm(X, IN[pm.saW[1][l]], nullptr, Y, M, Cin, Co, Co, Kp);
            k_stats<<<Co, 256, 0, stream>>>(Y, M, Co, mean, rstd);
            if (l < 2) {
                int Kpn = g_kpad[1][l + 1];
                long t2 = (long)M * Kpn;
                k_norm_relu_cast<<<GD(t2, 256), 256, 0, stream>>>(Y, mean, rstd,
                    IN[pm.saG[1][l]], IN[pm.saB[1][l]], Co, Kpn, X, t2);
            } else {
                long t2 = (long)BB * S * Co;
                k_norm_relu_maxpool<<<GD(t2, 256), 256, 0, stream>>>(Y, mean, rstd,
                    IN[pm.saG[1][l]], IN[pm.saB[1][l]], K, Co, l2f, t2);
            }
        }
    }

    // ---- SA3 (group-all over 128 points) ----
    {
        const int M = BB * 128;
        long tot = (long)M * g_kpad[2][0];
        k_concat_cast<<<GD(tot, 256), 256, 0, stream>>>(l2x, l2f, 256, g_kpad[2][0], X, tot);
        for (int l = 0; l < 3; l++) {
            int Cin = g_cin[2][l], Co = g_cout[2][l], Kp = g_kpad[2][l];
            gemm(X, IN[pm.saW[2][l]], nullptr, Y, M, Cin, Co, Co, Kp);
            k_stats<<<Co, 256, 0, stream>>>(Y, M, Co, mean, rstd);
            if (l < 2) {
                int Kpn = g_kpad[2][l + 1];
                long t2 = (long)M * Kpn;
                k_norm_relu_cast<<<GD(t2, 256), 256, 0, stream>>>(Y, mean, rstd,
                    IN[pm.saG[2][l]], IN[pm.saB[2][l]], Co, Kpn, X, t2);
            } else {
                long t2 = (long)BB * Co;   // groups = BB, K = 128 points
                k_norm_relu_maxpool<<<GD(t2, 256), 256, 0, stream>>>(Y, mean, rstd,
                    IN[pm.saG[2][l]], IN[pm.saB[2][l]], 128, Co, feat, t2);
            }
        }
    }

    // ---- FC head (biases before BN cancel; only fc3 bias matters) ----
    {
        k_cast_f16<<<GD((long)BB * 1024, 256), 256, 0, stream>>>(feat, X, (long)BB * 1024);
        // fc1 -> bn1 -> relu
        gemm(X, IN[pm.fc1w], nullptr, Y, BB, 1024, 512, 512, 1024);
        k_stats<<<512, 256, 0, stream>>>(Y, BB, 512, mean, rstd);
        {
            long t2 = (long)BB * 512;
            k_norm_relu_cast<<<GD(t2, 256), 256, 0, stream>>>(Y, mean, rstd,
                IN[pm.bn1g], IN[pm.bn1b], 512, 512, X, t2);
        }
        // fc2 -> bn2 -> relu
        gemm(X, IN[pm.fc2w], nullptr, Y, BB, 512, 256, 256, 512);
        k_stats<<<256, 256, 0, stream>>>(Y, BB, 256, mean, rstd);
        {
            long t2 = (long)BB * 256;
            k_norm_relu_cast<<<GD(t2, 256), 256, 0, stream>>>(Y, mean, rstd,
                IN[pm.bn2g], IN[pm.bn2b], 256, 256, X, t2);
        }
        // fc3 + bias -> d_out [32, 40]  (Npad = 64: multiple of 32, zero-padded weight rows)
        gemm(X, IN[pm.fc3w], IN[pm.fc3b], (float*)d_out, BB, 256, 40, 64, 256);
    }
}
